// StrongResidualGAT_86930138071435
// MI455X (gfx1250) — compile-verified, matrix operand order checked by reading, not run on hardware
//
#include <hip/hip_runtime.h>
#include <cstdint>

typedef __attribute__((ext_vector_type(16))) _Float16 v16h;
typedef __attribute__((ext_vector_type(8)))  float    v8f;

#define HEADS 4
#define HID 16
#define OUT_DIM 64
#define IN_CH 128
#define NUM_CLASSES 10

__device__ __forceinline__ float lrelu(float v) { return v > 0.f ? v : 0.2f * v; }

// Order-preserving float<->uint encoding so we can use hardware atomic_max_u32.
__device__ __forceinline__ unsigned fenc(float f) {
    unsigned u = __float_as_uint(f);
    return (u & 0x80000000u) ? ~u : (u | 0x80000000u);
}
__device__ __forceinline__ float fdec(unsigned e) {
    unsigned u = (e & 0x80000000u) ? (e ^ 0x80000000u) : ~e;
    return __uint_as_float(u);
}

// ---------------------------------------------------------------------------
// Pack W_gat (n-tiles 0..3) and W_res (n-tiles 4..7) into WMMA B lane layout,
// stored per-lane-contiguous so a B tile is two b128 loads per lane:
//   Bp[ ((kt*8+nt)*32 + lane)*8 + v ] = half2( W[K,n], W[K+1,n] )
//   K = kt*32 + (v/4)*16 + (lane/16)*8 + (v%4)*2,  n = ntile_base + lane%16
// ---------------------------------------------------------------------------
__global__ void k_pack_b(const float* __restrict__ Wg, const float* __restrict__ Wr,
                         unsigned* __restrict__ Bp) {
    int tid = blockIdx.x * blockDim.x + threadIdx.x;
    if (tid >= 4 * 8 * 32 * 8) return;
    int vv = tid & 7;
    int L  = (tid >> 3) & 31;
    int nt = (tid >> 8) & 7;
    int kt = tid >> 11;
    int half = L >> 4, n16 = L & 15;
    int K = kt * 32 + ((vv >> 2) << 4) + (half << 3) + ((vv & 3) << 1);
    const float* W = (nt < 4) ? Wg : Wr;
    int col = ((nt & 3) << 4) + n16;
    union HU { _Float16 h; unsigned short u; } h0, h1;
    h0.h = (_Float16)W[K * OUT_DIM + col];
    h1.h = (_Float16)W[(K + 1) * OUT_DIM + col];
    Bp[tid] = (unsigned)h0.u | ((unsigned)h1.u << 16);
}

// ---------------------------------------------------------------------------
// Dual GEMM: h = x @ W_gat, res = x @ W_res.  One wave per 16-row tile.
// A tile (16x32 f16) built on the fly from f32 x via b128 loads;
// 8 C accumulators (16x16 f32): 4 for h, 4 for res.
// ---------------------------------------------------------------------------
__global__ void k_gemm(const float* __restrict__ x, const unsigned* __restrict__ Bp,
                       float* __restrict__ hbuf, float* __restrict__ resbuf, int ntiles) {
    int wave = threadIdx.x >> 5;
    int rt = blockIdx.x * (blockDim.x >> 5) + wave;
    if (rt >= ntiles) return;
    int L = threadIdx.x & 31;
    int m = L & 15, half = L >> 4;
    const float* xrow = x + (size_t)(rt * 16 + m) * IN_CH;

    v8f acc[8];
#pragma unroll
    for (int i = 0; i < 8; ++i) acc[i] = (v8f){0.f, 0.f, 0.f, 0.f, 0.f, 0.f, 0.f, 0.f};

#pragma unroll
    for (int kt = 0; kt < 4; ++kt) {
        // Build A (16x32 f16): lane m = row; each lane reads 2 runs of 8 floats.
        v16h a;
#pragma unroll
        for (int g = 0; g < 2; ++g) {
            const float* base = xrow + kt * 32 + g * 16 + half * 8;
            const float4 x0 = *(const float4*)(base);
            const float4 x1 = *(const float4*)(base + 4);
            a[8 * g + 0] = (_Float16)x0.x;
            a[8 * g + 1] = (_Float16)x0.y;
            a[8 * g + 2] = (_Float16)x0.z;
            a[8 * g + 3] = (_Float16)x0.w;
            a[8 * g + 4] = (_Float16)x1.x;
            a[8 * g + 5] = (_Float16)x1.y;
            a[8 * g + 6] = (_Float16)x1.z;
            a[8 * g + 7] = (_Float16)x1.w;
        }
#pragma unroll
        for (int nt = 0; nt < 8; ++nt) {
            union { v16h vec; uint4 q[2]; } bu;
            const uint4* bp = (const uint4*)(Bp + (((kt * 8 + nt) * 32) + L) * 8);
            bu.q[0] = bp[0];
            bu.q[1] = bp[1];
            acc[nt] = __builtin_amdgcn_wmma_f32_16x16x32_f16(
                false, a, false, bu.vec, (short)0, acc[nt], false, false);
        }
    }
    // C layout: VGPR r -> M = r (lanes 0-15) / r+8 (lanes 16-31), N = lane%16.
#pragma unroll
    for (int nt = 0; nt < 8; ++nt) {
        float* dst = (nt < 4) ? hbuf : resbuf;
        int colb = (nt & 3) * 16;
#pragma unroll
        for (int r = 0; r < 8; ++r) {
            int orow = rt * 16 + half * 8 + r;
            dst[(size_t)orow * OUT_DIM + colb + m] = acc[nt][r];
        }
    }
}

// Per (node, head): attention logits + self-loop max seed.
__global__ void k_node_att(const float* __restrict__ hbuf, const float* __restrict__ att_s,
                           const float* __restrict__ att_d, float* __restrict__ a_src,
                           float* __restrict__ a_dst, unsigned* __restrict__ menc, int n) {
    int i = blockIdx.x * blockDim.x + threadIdx.x;
    if (i >= n * HEADS) return;
    int node = i >> 2, hd = i & 3;
    const float* hrow = hbuf + (size_t)node * OUT_DIM + hd * HID;
    float s = 0.f, d = 0.f;
#pragma unroll
    for (int c = 0; c < HID; ++c) {
        float hv = hrow[c];
        s += hv * att_s[hd * HID + c];
        d += hv * att_d[hd * HID + c];
    }
    a_src[i] = s;
    a_dst[i] = d;
    menc[i] = fenc(lrelu(s + d));  // seed segment-max with self-loop logit
}

__global__ void k_edge_max(const int* __restrict__ ei, const float* __restrict__ a_src,
                           const float* __restrict__ a_dst, unsigned* __restrict__ menc, int E) {
    int e = blockIdx.x * blockDim.x + threadIdx.x;
    if (e >= E) return;
    int s = ei[e], d = ei[E + e];
#pragma unroll
    for (int hd = 0; hd < HEADS; ++hd) {
        float ev = lrelu(a_src[s * 4 + hd] + a_dst[d * 4 + hd]);
        atomicMax(&menc[d * 4 + hd], fenc(ev));
    }
}

__global__ void k_node_mdec(const unsigned* __restrict__ menc, const float* __restrict__ a_src,
                            const float* __restrict__ a_dst, float* __restrict__ mf,
                            float* __restrict__ denom, int n) {
    int i = blockIdx.x * blockDim.x + threadIdx.x;
    if (i >= n * HEADS) return;
    float mm = fdec(menc[i]);
    mf[i] = mm;
    denom[i] = __expf(lrelu(a_src[i] + a_dst[i]) - mm);  // self-loop term
}

__global__ void k_edge_sum(const int* __restrict__ ei, const float* __restrict__ a_src,
                           const float* __restrict__ a_dst, const float* __restrict__ mf,
                           float* __restrict__ denom, int E) {
    int t = blockIdx.x * blockDim.x + threadIdx.x;
    if (t >= E * 4) return;
    int e = t >> 2, hd = t & 3;
    int s = ei[e], d = ei[E + e];
    float ev = lrelu(a_src[s * 4 + hd] + a_dst[d * 4 + hd]);
    atomicAdd(&denom[d * 4 + hd], __expf(ev - mf[d * 4 + hd]));
}

// 16 lanes per edge; each lane handles 4 channels (one head per 4 lanes).
__global__ void k_edge_agg(const int* __restrict__ ei, const float* __restrict__ a_src,
                           const float* __restrict__ a_dst, const float* __restrict__ mf,
                           const float* __restrict__ denom, const float* __restrict__ hbuf,
                           float* __restrict__ agg, int E) {
    long long t = (long long)blockIdx.x * blockDim.x + threadIdx.x;
    if (t >= (long long)E * 16) return;
    int e = (int)(t >> 4), lane = (int)(t & 15);
    int s = ei[e], d = ei[E + e];
    int hd = lane >> 2;
    int idx = d * 4 + hd;
    float ev = lrelu(a_src[s * 4 + hd] + a_dst[idx]);
    float alpha = __expf(ev - mf[idx]) / (denom[idx] + 1e-16f);
    const float4 hv = *(const float4*)(hbuf + (size_t)s * OUT_DIM + lane * 4);
    float* dst = agg + (size_t)d * OUT_DIM + lane * 4;
    atomicAdd(dst + 0, alpha * hv.x);
    atomicAdd(dst + 1, alpha * hv.y);
    atomicAdd(dst + 2, alpha * hv.z);
    atomicAdd(dst + 3, alpha * hv.w);
}

// Self-loop message + biases + residual + BN(eval) + ReLU + [64x10] classifier.
__global__ void k_epilogue(const float* __restrict__ agg, const float* __restrict__ hbuf,
                           const float* __restrict__ resbuf, const float* __restrict__ a_src,
                           const float* __restrict__ a_dst, const float* __restrict__ mf,
                           const float* __restrict__ denom, const float* __restrict__ b_gat,
                           const float* __restrict__ b_res, const float* __restrict__ gamma,
                           const float* __restrict__ beta, const float* __restrict__ rmean,
                           const float* __restrict__ rvar, const float* __restrict__ Wc,
                           const float* __restrict__ b_cls, float* __restrict__ out, int n) {
    int node = blockIdx.x * blockDim.x + threadIdx.x;
    if (node >= n) return;
    float alphaS[HEADS];
#pragma unroll
    for (int hd = 0; hd < HEADS; ++hd) {
        int i = node * 4 + hd;
        float ev = lrelu(a_src[i] + a_dst[i]);
        alphaS[hd] = __expf(ev - mf[i]) / (denom[i] + 1e-16f);
    }
    float lg[NUM_CLASSES];
#pragma unroll
    for (int k = 0; k < NUM_CLASSES; ++k) lg[k] = b_cls[k];
    for (int c = 0; c < OUT_DIM; ++c) {
        int hd = c >> 4;
        size_t off = (size_t)node * OUT_DIM + c;
        float val = agg[off] + alphaS[hd] * hbuf[off] + b_gat[c] + resbuf[off] + b_res[c];
        val = (val - rmean[c]) * rsqrtf(rvar[c] + 1e-5f) * gamma[c] + beta[c];
        val = fmaxf(val, 0.f);
#pragma unroll
        for (int k = 0; k < NUM_CLASSES; ++k) lg[k] += val * Wc[c * NUM_CLASSES + k];
    }
#pragma unroll
    for (int k = 0; k < NUM_CLASSES; ++k) out[(size_t)node * NUM_CLASSES + k] = lg[k];
}

extern "C" void kernel_launch(void* const* d_in, const int* in_sizes, int n_in,
                              void* d_out, int out_size, void* d_ws, size_t ws_size,
                              hipStream_t stream) {
    (void)n_in; (void)out_size; (void)ws_size;
    const float* x     = (const float*)d_in[0];
    const int*   ei    = (const int*)d_in[1];
    const float* Wg    = (const float*)d_in[2];
    const float* att_s = (const float*)d_in[3];
    const float* att_d = (const float*)d_in[4];
    const float* b_gat = (const float*)d_in[5];
    const float* Wr    = (const float*)d_in[6];
    const float* b_res = (const float*)d_in[7];
    const float* gamma = (const float*)d_in[8];
    const float* beta  = (const float*)d_in[9];
    const float* rmean = (const float*)d_in[10];
    const float* rvar  = (const float*)d_in[11];
    const float* Wc    = (const float*)d_in[12];
    const float* b_cls = (const float*)d_in[13];
    float* out = (float*)d_out;

    const int N = in_sizes[0] / IN_CH;
    const int E = in_sizes[1] / 2;

    char* ws = (char*)d_ws;
    size_t off = 0;
    auto alloc = [&](size_t bytes) -> char* {
        char* p = ws + off;
        off += (bytes + 255) & ~(size_t)255;
        return p;
    };
    float*    hbuf   = (float*)alloc((size_t)N * OUT_DIM * 4);
    float*    resbuf = (float*)alloc((size_t)N * OUT_DIM * 4);
    float*    agg    = (float*)alloc((size_t)N * OUT_DIM * 4);
    float*    a_src  = (float*)alloc((size_t)N * HEADS * 4);
    float*    a_dst  = (float*)alloc((size_t)N * HEADS * 4);
    unsigned* menc   = (unsigned*)alloc((size_t)N * HEADS * 4);
    float*    mf     = (float*)alloc((size_t)N * HEADS * 4);
    float*    denom  = (float*)alloc((size_t)N * HEADS * 4);
    unsigned* Bp     = (unsigned*)alloc(8192 * 4);

    hipMemsetAsync(agg, 0, (size_t)N * OUT_DIM * 4, stream);
    k_pack_b<<<32, 256, 0, stream>>>(Wg, Wr, Bp);

    int ntiles = (N + 15) / 16;
    k_gemm<<<(ntiles + 3) / 4, 128, 0, stream>>>(x, Bp, hbuf, resbuf, ntiles);

    k_node_att<<<(N * 4 + 255) / 256, 256, 0, stream>>>(hbuf, att_s, att_d, a_src, a_dst, menc, N);
    k_edge_max<<<(E + 255) / 256, 256, 0, stream>>>(ei, a_src, a_dst, menc, E);
    k_node_mdec<<<(N * 4 + 255) / 256, 256, 0, stream>>>(menc, a_src, a_dst, mf, denom, N);
    k_edge_sum<<<(E * 4 + 255) / 256, 256, 0, stream>>>(ei, a_src, a_dst, mf, denom, E);

    long long aggThreads = (long long)E * 16;
    k_edge_agg<<<(unsigned)((aggThreads + 255) / 256), 256, 0, stream>>>(
        ei, a_src, a_dst, mf, denom, hbuf, agg, E);

    k_epilogue<<<(N + 255) / 256, 256, 0, stream>>>(agg, hbuf, resbuf, a_src, a_dst, mf, denom,
                                                    b_gat, b_res, gamma, beta, rmean, rvar,
                                                    Wc, b_cls, out, N);
}
